// CellularAutomatonDecoder_11407433138222
// MI455X (gfx1250) — matrix operand order repaired and checked by reading, not compiled, not run on hardware
//
#include <hip/hip_runtime.h>
#include <hip/hip_bf16.h>
#include <math.h>

typedef __attribute__((ext_vector_type(16))) _Float16 v16h;
typedef __attribute__((ext_vector_type(8)))  _Float16 v8h;
typedef __attribute__((ext_vector_type(8)))  float    v8f;

#define BB 2048
#define TT 32
#define DD 128
#define VV 256
#define NEV 8

__device__ __forceinline__ float gelu_exact(float x){
    return 0.5f * x * (1.0f + erff(x * 0.70710678118654752f));
}

// ---------- prep 1: rule-network bias -> const_bias (256 floats) ----------
__global__ __launch_bounds__(256)
void prep_bias_kernel(const float* __restrict__ c_states,
                      const float* __restrict__ Wc1, const float* __restrict__ bc1,
                      const float* __restrict__ Wc2, const float* __restrict__ bc2,
                      const float* __restrict__ W1,  const float* __restrict__ b1,
                      float* __restrict__ constb){
    __shared__ float pooled[DD];
    __shared__ float h1[2*DD];
    __shared__ float rb[DD];
    const int tid = threadIdx.x;
    if (tid < DD)
        pooled[tid] = 0.25f*(c_states[tid] + c_states[DD+tid] +
                             c_states[2*DD+tid] + c_states[3*DD+tid]);
    __syncthreads();
    {
        float s = bc1[tid];
        for (int j = 0; j < DD; ++j) s += pooled[j]*Wc1[j*(2*DD)+tid];
        h1[tid] = gelu_exact(s);
    }
    __syncthreads();
    if (tid < DD){
        float s = bc2[tid];
        for (int n = 0; n < 2*DD; ++n) s += h1[n]*Wc2[n*DD+tid];
        rb[tid] = s;
    }
    __syncthreads();
    {
        // const_bias = rule_bias @ W1b + b1 ; W1b = W1 rows [384, 512)
        float s = b1[tid];
        for (int d = 0; d < DD; ++d) s += rb[d]*W1[(3*DD+d)*(2*DD)+tid];
        constb[tid] = s;
    }
}

// ---------- prep 2: pack weights into per-lane WMMA B-fragment order ----------
// B fragment as packed here (contiguous k, matching load_afrag): lane L holds N = L&15, halves h -> K = (L>>4)*16 + h (+32*kk)
__global__ __launch_bounds__(256)
void prep_pack_kernel(const float* __restrict__ W1, const float* __restrict__ W2,
                      const float* __restrict__ head_w,
                      _Float16* __restrict__ wcatF, _Float16* __restrict__ w2F,
                      _Float16* __restrict__ headF){
    const int idx = blockIdx.x*256 + threadIdx.x;
    if (idx < 98304){                       // Wcat = W1 rows [0,384): 16 nt x 12 kk x 32 lane x 16 h
        int u = idx;
        int h = u & 15, lane = (u>>4)&31;
        int kk = (u>>9) % 12;
        int nt = u / (12*32*16);
        int K = kk*32 + ((lane>>4)<<4) + h;
        int N = nt*16 + (lane&15);
        wcatF[idx] = (_Float16)W1[K*(2*DD) + N];
    } else if (idx < 98304 + 32768){        // W2 (256x128): 8 nt x 8 kk
        int u = idx - 98304;
        int h = u & 15, lane = (u>>4)&31;
        int kk = (u>>9)&7, nt = u>>12;
        int K = kk*32 + ((lane>>4)<<4) + h;
        int N = nt*16 + (lane&15);
        w2F[u] = (_Float16)W2[K*DD + N];
    } else if (idx < 98304 + 65536){        // head_w (128x256): 16 nt x 4 kk
        int u = idx - 98304 - 32768;
        int h = u & 15, lane = (u>>4)&31;
        int kk = (u>>9)&3, nt = u>>11;
        int K = kk*32 + ((lane>>4)<<4) + h;
        int N = nt*16 + (lane&15);
        headF[u] = (_Float16)head_w[K*VV + N];
    }
}

// Build v16h A-fragment from two contiguous 16B LDS chunks (k order used here, not AMD's table order:
// lane group g=lane>>4 selects K-halves; halves 0-15 -> K=base..base+15, contiguous)
__device__ __forceinline__ v16h load_afrag(const _Float16* rowPtr, int col0){
    v8h a0 = *(const v8h*)(rowPtr + col0);
    v8h a1 = *(const v8h*)(rowPtr + col0 + 8);
    v16h a;
    #pragma unroll
    for (int i = 0; i < 8; ++i){ a[i] = a0[i]; a[8+i] = a1[i]; }
    return a;
}

// ---------- main: one block per batch row; cells live in LDS across all 8 steps ----------
__global__ __launch_bounds__(256)
void ca_evolve_kernel(const int* __restrict__ tokens, const float* __restrict__ embed,
                      const float* __restrict__ pos_embed,
                      const float* __restrict__ b2, const float* __restrict__ alpha,
                      const float* __restrict__ ln_g, const float* __restrict__ ln_b,
                      const float* __restrict__ constb,
                      const _Float16* __restrict__ wcatF, const _Float16* __restrict__ w2F,
                      const _Float16* __restrict__ headF,
                      float* __restrict__ out){
    __shared__ __align__(32) float    cellsF[TT][DD];
    __shared__ __align__(32) _Float16 cellsH[TT][DD];
    __shared__ __align__(32) _Float16 preH[TT][2*DD];
    __shared__ float cb[2*DD];

    const int tid  = threadIdx.x;
    const int lane = tid & 31;
    const int wid  = tid >> 5;
    const int b    = blockIdx.x;
    const int g    = lane >> 4;   // lane group
    const int lr   = lane & 15;   // row/col within group

    // cells = embed[tokens] + pos_embed
    for (int i = tid; i < TT*DD; i += 256){
        int t = i >> 7, d = i & 127;
        int tok = tokens[b*TT + t];
        float v = embed[tok*DD + d] + pos_embed[t*DD + d];
        cellsF[t][d] = v;
        cellsH[t][d] = (_Float16)v;
    }
    if (tid < 2*DD) cb[tid] = constb[tid];
    const float aa = 1.0f/(1.0f + expf(-alpha[0]));
    __syncthreads();

    for (int step = 0; step < NEV; ++step){
        // GEMM1: pre(32x256) = [cells|left|right](32x384) @ Wcat + cb ; gelu -> preH (f16)
        for (int tile = wid; tile < 32; tile += 8){
            const int mt = tile >> 4, nt = tile & 15;
            v8f acc = {};
            for (int kk = 0; kk < 12; ++kk){
                const int sel  = kk >> 2;            // 0=center,1=left(t-1),2=right(t+1)
                const int kloc = (kk & 3) * 32;
                const int t0   = mt*16 + lr;
                const int trow = (sel == 0) ? t0
                               : (sel == 1) ? ((t0 + TT - 1) & (TT - 1))
                                            : ((t0 + 1) & (TT - 1));
                v16h afrag = load_afrag(&cellsH[trow][0], kloc + g*16);
                const v16h* pb = (const v16h*)wcatF + ((nt*12 + kk)*32 + lane);
                if (kk < 11) __builtin_prefetch((const void*)(pb + 32), 0, 1);
                v16h bfrag = *pb;
                acc = __builtin_amdgcn_wmma_f32_16x16x32_f16(false, afrag, false, bfrag,
                                                             (short)0, acc, false, false);
            }
            const int N = nt*16 + lr;
            const float bias = cb[N];
            #pragma unroll
            for (int r = 0; r < 8; ++r){
                int M = mt*16 + g*8 + r;
                preH[M][N] = (_Float16)gelu_exact(acc[r] + bias);
            }
        }
        __syncthreads();

        // GEMM2: new(32x128) = preH(32x256) @ W2 + b2 ; tanh ; blend into cells
        for (int tile = wid; tile < 16; tile += 8){
            const int mt = tile >> 3, nt = tile & 7;
            v8f acc = {};
            for (int kk = 0; kk < 8; ++kk){
                const int trow = mt*16 + lr;
                v16h afrag = load_afrag(&preH[trow][0], kk*32 + g*16);
                const v16h* pb = (const v16h*)w2F + ((nt*8 + kk)*32 + lane);
                v16h bfrag = *pb;
                acc = __builtin_amdgcn_wmma_f32_16x16x32_f16(false, afrag, false, bfrag,
                                                             (short)0, acc, false, false);
            }
            const int N = nt*16 + lr;
            const float bias2 = b2[N];
            #pragma unroll
            for (int r = 0; r < 8; ++r){
                int M = mt*16 + g*8 + r;
                float nv  = tanhf(acc[r] + bias2);
                float res = aa*cellsF[M][N] + (1.0f - aa)*nv;
                cellsF[M][N] = res;
                cellsH[M][N] = (_Float16)res;
            }
        }
        __syncthreads();
    }

    // LayerNorm over D: 32 threads, one T-row each
    if (tid < TT){
        float mu = 0.f;
        for (int d = 0; d < DD; ++d) mu += cellsF[tid][d];
        mu *= (1.0f/DD);
        float var = 0.f;
        for (int d = 0; d < DD; ++d){ float x = cellsF[tid][d] - mu; var += x*x; }
        var *= (1.0f/DD);
        const float rs = rsqrtf(var + 1e-5f);
        for (int d = 0; d < DD; ++d){
            float x = (cellsF[tid][d] - mu)*rs*ln_g[d] + ln_b[d];
            cellsH[tid][d] = (_Float16)x;
        }
    }
    __syncthreads();

    // Head GEMM: out(32x256) = cellsH(32x128) @ head_w
    for (int tile = wid; tile < 32; tile += 8){
        const int mt = tile >> 4, nt = tile & 15;
        v8f acc = {};
        for (int kk = 0; kk < 4; ++kk){
            const int trow = mt*16 + lr;
            v16h afrag = load_afrag(&cellsH[trow][0], kk*32 + g*16);
            const v16h* pb = (const v16h*)headF + ((nt*4 + kk)*32 + lane);
            v16h bfrag = *pb;
            acc = __builtin_amdgcn_wmma_f32_16x16x32_f16(false, afrag, false, bfrag,
                                                         (short)0, acc, false, false);
        }
        const int N = nt*16 + lr;
        #pragma unroll
        for (int r = 0; r < 8; ++r){
            int M = mt*16 + g*8 + r;
            out[(b*TT + M)*VV + N] = acc[r];
        }
    }
}

extern "C" void kernel_launch(void* const* d_in, const int* in_sizes, int n_in,
                              void* d_out, int out_size, void* d_ws, size_t ws_size,
                              hipStream_t stream){
    const int*   tokens    = (const int*)  d_in[0];
    const float* c_states  = (const float*)d_in[1];
    const float* embed     = (const float*)d_in[2];
    const float* pos_embed = (const float*)d_in[3];
    const float* W1        = (const float*)d_in[4];
    const float* b1        = (const float*)d_in[5];
    const float* W2        = (const float*)d_in[6];
    const float* b2        = (const float*)d_in[7];
    const float* Wc1       = (const float*)d_in[8];
    const float* bc1       = (const float*)d_in[9];
    const float* Wc2       = (const float*)d_in[10];
    const float* bc2       = (const float*)d_in[11];
    const float* alpha     = (const float*)d_in[12];
    const float* ln_g      = (const float*)d_in[13];
    const float* ln_b      = (const float*)d_in[14];
    const float* head_w    = (const float*)d_in[15];
    float* out = (float*)d_out;

    char* ws = (char*)d_ws;
    float*    constb = (float*)ws;                                   // 1 KB
    _Float16* wcatF  = (_Float16*)(ws + 1024);                       // 192 KB
    _Float16* w2F    = (_Float16*)(ws + 1024 + 196608);              // 64 KB
    _Float16* headF  = (_Float16*)(ws + 1024 + 196608 + 65536);      // 64 KB

    prep_bias_kernel<<<1, 256, 0, stream>>>(c_states, Wc1, bc1, Wc2, bc2, W1, b1, constb);
    const int packTotal = 98304 + 32768 + 32768;
    prep_pack_kernel<<<(packTotal + 255)/256, 256, 0, stream>>>(W1, W2, head_w,
                                                                wcatF, w2F, headF);
    ca_evolve_kernel<<<BB, 256, 0, stream>>>(tokens, embed, pos_embed, b2, alpha,
                                             ln_g, ln_b, constb, wcatF, w2F, headF, out);
}
